// DeepseekV3DecoderLayer_52673478918594
// MI455X (gfx1250) — compile-verified, matrix operand order checked by reading, not run on hardware
//
#include <hip/hip_runtime.h>
#include <hip/hip_bf16.h>
#include <math.h>

typedef __bf16 bf16_t;
typedef __attribute__((ext_vector_type(16))) __bf16 v16bf;
typedef __attribute__((ext_vector_type(8)))  __bf16 v8bf;
typedef __attribute__((ext_vector_type(8)))  float  v8f;

#define DEV __device__ __forceinline__

// ---- model dims (DeepseekV3 MLA layer, B=2 S=1024) ----
#define DIM_H    2048
#define DIM_NH   16
#define DIM_QLR  1536
#define DIM_KVLR 512
#define DIM_NOPE 128
#define DIM_ROPE 64
#define DIM_VD   128
#define DIM_QHD  192
#define DIM_I    8192
#define DIM_B    2
#define DIM_S    1024
#define ROWS     (DIM_B * DIM_S)

// trivial 16-byte vector for LDS/global block copies (HIP uint4 has ctors)
struct alignas(16) U4 { unsigned x, y, z, w; };

DEV v8f wmma_bf16(v16bf a, v16bf b, v8f c) {
  // (neg_a, A, neg_b, B, c_mod, C, reuse_a, reuse_b)
  return __builtin_amdgcn_wmma_f32_16x16x32_bf16(false, a, false, b, (short)0, c,
                                                 false, false);
}

// Load one 16x32 bf16 fragment in the documented A/B per-lane layout:
// lane l: row = l&15, k-base = (l<16?0:8); VGPR0..3 = k pairs {kb..kb+7},
// VGPR4..7 = {kb+16..kb+23}.  base points at element (0,0); stride in elements.
DEV v16bf load_frag(const bf16_t* base, int stride) {
  const int l = (int)(threadIdx.x & 31);
  const bf16_t* p = base + (l & 15) * stride + ((l < 16) ? 0 : 8);
  v8bf lo = *(const v8bf*)(p);
  v8bf hi = *(const v8bf*)(p + 16);
  return __builtin_shufflevector(lo, hi, 0, 1, 2, 3, 4, 5, 6, 7,
                                 8, 9, 10, 11, 12, 13, 14, 15);
}

// generic LDS pointer -> 32-bit LDS offset (generic LDS addr keeps offset in [31:0])
DEV unsigned lds_off(const void* p) { return (unsigned)(uintptr_t)p; }

DEV void async_b128_to_lds(unsigned lds, unsigned goffB, const void* sbase) {
  // GVS mode: mem = SADDR(64) + VADDR(i32) + IOFFSET; VDST = per-lane LDS offset.
  asm volatile("global_load_async_to_lds_b128 %0, %1, %2"
               :: "v"(lds), "v"(goffB), "s"(sbase) : "memory");
}

DEV void wait_async0() { asm volatile("s_wait_asynccnt 0" ::: "memory"); }

// ===================== GEMM: out = A(bf16)[M,K] @ Bw(f32)[K,N] =====================
// mode 0: outF = acc            mode 1: outF = acc + res
// mode 2: outB = bf16(acc)      mode 3: outB = bf16(silu(aux) * acc)
// Double-buffered: A tile via async global->LDS DMA, B tile via f32 loads + cvt.
#define BM 128
#define BN 128
#define BK 32
#define LSA 40  // padded LDS stride (bf16 elements)

__global__ __launch_bounds__(256) void gemm_bf16_kernel(
    const bf16_t* __restrict__ A, const float* __restrict__ Bw,
    int M, int N, int K, int mode,
    float* __restrict__ outF, bf16_t* __restrict__ outB,
    const float* __restrict__ res, const float* __restrict__ aux) {
  __shared__ __align__(16) bf16_t As[2][BM * LSA];
  __shared__ __align__(16) bf16_t Bs[2][BN * LSA];  // transposed: Bs[n][k]

  const int tid = (int)threadIdx.x;
  const int wid = tid >> 5;
  const int wm = wid & 3;   // 4 wave-rows of 32
  const int wn = wid >> 2;  // 2 wave-cols of 64
  const int m0 = (int)blockIdx.y * BM;
  const int n0 = (int)blockIdx.x * BN;
  const bf16_t* Abase = A + (size_t)m0 * K;

  const v8f vz = {0.f, 0.f, 0.f, 0.f, 0.f, 0.f, 0.f, 0.f};
  v8f acc[2][4];
  for (int i = 0; i < 2; ++i)
    for (int j = 0; j < 4; ++j) acc[i][j] = vz;

  // --- staging helpers ---
  auto stageA = [&](int buf, int k0) {  // 128x32 bf16 tile, 512 x 16B async DMAs
#pragma unroll
    for (int it = 0; it < 2; ++it) {
      int slot = tid + it * 256;
      int r = slot >> 2;
      int kq = (slot & 3) * 8;
      async_b128_to_lds(lds_off(&As[buf][r * LSA + kq]),
                        (unsigned)(((unsigned)r * (unsigned)K +
                                    (unsigned)(k0 + kq)) * 2u),
                        Abase);
    }
  };
  float4 br[4];
  auto loadB = [&](int k0) {  // 32x128 f32 tile into registers (branchless guard)
#pragma unroll
    for (int it = 0; it < 4; ++it) {
      int k = (tid >> 5) + it * 8;
      int gn = n0 + (tid & 31) * 4;
      br[it] = make_float4(0.f, 0.f, 0.f, 0.f);
      if (gn < N)  // N is always a multiple of 4 and gn a multiple of 4
        br[it] = *(const float4*)&Bw[(size_t)(k0 + k) * N + gn];
    }
  };
  auto storeB = [&](int buf) {  // cvt f32->bf16, store transposed Bs[n][k]
#pragma unroll
    for (int it = 0; it < 4; ++it) {
      int k = (tid >> 5) + it * 8;
      int n4 = (tid & 31) * 4;
      Bs[buf][(n4 + 0) * LSA + k] = (bf16_t)br[it].x;
      Bs[buf][(n4 + 1) * LSA + k] = (bf16_t)br[it].y;
      Bs[buf][(n4 + 2) * LSA + k] = (bf16_t)br[it].z;
      Bs[buf][(n4 + 3) * LSA + k] = (bf16_t)br[it].w;
    }
  };

  // --- prologue: tile 0 ---
  stageA(0, 0);
  loadB(0);
  storeB(0);
  wait_async0();
  __syncthreads();

  const int nk = K / BK;
  for (int kt = 0; kt < nk; ++kt) {
    const int cur = kt & 1, nxt = cur ^ 1;
    const bool more = (kt + 1) < nk;
    if (more) {
      stageA(nxt, (kt + 1) * BK);  // async DMA overlaps with compute below
      loadB((kt + 1) * BK);        // f32 weight loads in flight during WMMAs
    }
    // compute current tile
    v16bf bfrag[4];
#pragma unroll
    for (int j = 0; j < 4; ++j)
      bfrag[j] = load_frag(&Bs[cur][(wn * 64 + j * 16) * LSA], LSA);
#pragma unroll
    for (int i = 0; i < 2; ++i) {
      v16bf afrag = load_frag(&As[cur][(wm * 32 + i * 16) * LSA], LSA);
#pragma unroll
      for (int j = 0; j < 4; ++j)
        acc[i][j] = wmma_bf16(afrag, bfrag[j], acc[i][j]);
    }
    if (more) storeB(nxt);  // writes go to the idle buffer
    wait_async0();
    __syncthreads();
  }

  // Epilogue. C layout: lane l -> col = frag_n + (l&15); rows = frag_m + (l<16?0:8)+e.
  const int lane = tid & 31;
  const int cif = lane & 15;
  const int rh = (lane < 16) ? 0 : 8;
#pragma unroll
  for (int i = 0; i < 2; ++i) {
#pragma unroll
    for (int j = 0; j < 4; ++j) {
      int col = n0 + wn * 64 + j * 16 + cif;
      if (col >= N) continue;
#pragma unroll
      for (int e = 0; e < 8; ++e) {
        size_t row = (size_t)(m0 + wm * 32 + i * 16 + rh + e);
        size_t idx = row * (size_t)N + (size_t)col;
        float v = acc[i][j][e];
        if (mode == 0) {
          outF[idx] = v;
        } else if (mode == 1) {
          outF[idx] = v + res[idx];
        } else if (mode == 2) {
          outB[idx] = (bf16_t)v;
        } else {
          float g = aux[idx];
          float sg = g / (1.f + __expf(-g));  // silu(gate)
          outB[idx] = (bf16_t)(sg * v);
        }
      }
    }
  }
}

// ===================== RMSNorm: f32 row -> bf16 row =====================
__global__ __launch_bounds__(256) void rmsnorm_kernel(
    const float* __restrict__ x, int ldx, const float* __restrict__ w,
    bf16_t* __restrict__ out, int ldo, int n) {
  const int row = (int)blockIdx.x;
  const int tid = (int)threadIdx.x;
  const float* xr = x + (size_t)row * ldx;
  float ss = 0.f;
  for (int c = tid; c < n; c += 256) {
    float v = xr[c];
    ss += v * v;
  }
#pragma unroll
  for (int m = 16; m >= 1; m >>= 1) ss += __shfl_xor(ss, m, 32);
  __shared__ float red[8];
  if ((tid & 31) == 0) red[tid >> 5] = ss;
  __syncthreads();
  float tot = 0.f;
#pragma unroll
  for (int i = 0; i < 8; ++i) tot += red[i];
  float inv = rsqrtf(tot / (float)n + 1e-6f);
  bf16_t* orow = out + (size_t)row * ldo;
  for (int c = tid; c < n; c += 256) orow[c] = (bf16_t)(xr[c] * inv * w[c]);
}

// ===================== RoPE on q (interleaved), folds SCALE, writes [b,h,s,192] bf16 ===
__global__ __launch_bounds__(192) void rope_q_kernel(
    const float* __restrict__ q, const float* __restrict__ cosT,
    const float* __restrict__ sinT, const int* __restrict__ pos,
    bf16_t* __restrict__ Qbf, float scale) {
  const int row = (int)blockIdx.x;  // b*S + s
  const int h = (int)blockIdx.y;
  const int tid = (int)threadIdx.x;
  const int b = row / DIM_S;
  const int s = row - b * DIM_S;
  const float* qi = q + (size_t)row * (DIM_NH * DIM_QHD) + h * DIM_QHD;
  bf16_t* qo = Qbf + ((size_t)(b * DIM_NH + h) * DIM_S + s) * DIM_QHD;
  if (tid < DIM_NOPE) {
    qo[tid] = (bf16_t)(qi[tid] * scale);
  } else {
    int j = tid - DIM_NOPE;  // 0..63
    int k = j & 31;
    int p = pos[row];
    float x0 = qi[DIM_NOPE + 2 * k];
    float x1 = qi[DIM_NOPE + 2 * k + 1];
    float c = cosT[(size_t)p * DIM_ROPE + j];
    float sn = sinT[(size_t)p * DIM_ROPE + j];
    float v = (j < 32) ? (x0 * c - x1 * sn) : (x1 * c + x0 * sn);
    qo[tid] = (bf16_t)(v * scale);
  }
}

// ===================== RoPE on k_pe (last 64 of ckv) -> bf16 [rows,64] ======
__global__ __launch_bounds__(64) void rope_k_kernel(
    const float* __restrict__ ckv, const float* __restrict__ cosT,
    const float* __restrict__ sinT, const int* __restrict__ pos,
    bf16_t* __restrict__ kpe) {
  const int row = (int)blockIdx.x;
  const int j = (int)threadIdx.x;
  const int k = j & 31;
  const float* xr = ckv + (size_t)row * (DIM_KVLR + DIM_ROPE) + DIM_KVLR;
  int p = pos[row];
  float x0 = xr[2 * k], x1 = xr[2 * k + 1];
  float c = cosT[(size_t)p * DIM_ROPE + j];
  float sn = sinT[(size_t)p * DIM_ROPE + j];
  float v = (j < 32) ? (x0 * c - x1 * sn) : (x1 * c + x0 * sn);
  kpe[(size_t)row * DIM_ROPE + j] = (bf16_t)v;
}

// ===================== build K[b,h,s,192] (nope|rope) and V[b,h,s,128] bf16 ===
__global__ __launch_bounds__(192) void build_kv_kernel(
    const float* __restrict__ kv, const bf16_t* __restrict__ kpe,
    bf16_t* __restrict__ Kbf, bf16_t* __restrict__ Vbf) {
  const int row = (int)blockIdx.x;
  const int h = (int)blockIdx.y;
  const int tid = (int)threadIdx.x;
  const int b = row / DIM_S;
  const int s = row - b * DIM_S;
  size_t bh = (size_t)(b * DIM_NH + h) * DIM_S + s;
  const float* kvr = kv + (size_t)row * (DIM_NH * 256) + h * 256;
  if (tid < DIM_NOPE) {
    Kbf[bh * DIM_QHD + tid] = (bf16_t)kvr[tid];
    Vbf[bh * DIM_VD + tid] = (bf16_t)kvr[DIM_NOPE + tid];
  } else {
    Kbf[bh * DIM_QHD + tid] = kpe[(size_t)row * DIM_ROPE + (tid - DIM_NOPE)];
  }
}

// ===================== Flash attention (causal), WMMA QK^T and PV ============
#define SKLDS 200  // K tile stride (192 + 8)
#define SVLDS 40   // V^T tile stride (32 + 8)
#define SPLDS 40   // P tile stride (32 + 8)

__global__ __launch_bounds__(128) void flash_attn_kernel(
    const bf16_t* __restrict__ Q, const bf16_t* __restrict__ Kg,
    const bf16_t* __restrict__ Vg, const int* __restrict__ amask,
    bf16_t* __restrict__ O) {
  __shared__ __align__(16) bf16_t Kls[32 * SKLDS];     // K chunk [key][d]
  __shared__ __align__(16) bf16_t Vls[128 * SVLDS];    // V chunk transposed [d][key]
  __shared__ __align__(16) bf16_t Pls[4 * 16 * SPLDS]; // per-wave P bounce

  const int tid = (int)threadIdx.x;
  const int wid = tid >> 5;
  const int lane = tid & 31;
  const int q0 = (int)blockIdx.x * 64;
  const int h = (int)blockIdx.y;
  const int b = (int)blockIdx.z;
  const int qw = q0 + wid * 16;  // this wave's 16 q rows
  const size_t bh = (size_t)(b * DIM_NH + h) * DIM_S;
  const bf16_t* Kbase = Kg + bh * DIM_QHD;

  // Preload Q fragments (16 rows x 192 = 6 A-frags), straight from global.
  v16bf qf[6];
  {
    const bf16_t* qbase = Q + (bh + qw) * DIM_QHD;
#pragma unroll
    for (int d = 0; d < 6; ++d) qf[d] = load_frag(qbase + d * 32, DIM_QHD);
  }

  const int rh = (lane < 16) ? 0 : 8;
  const int lc = lane & 15;
  const v8f vz = {0.f, 0.f, 0.f, 0.f, 0.f, 0.f, 0.f, 0.f};
  float mrow[8], lrow[8];
  v8f oacc[8];
#pragma unroll
  for (int e = 0; e < 8; ++e) { mrow[e] = -3.0e38f; lrow[e] = 0.f; }
#pragma unroll
  for (int nf = 0; nf < 8; ++nf) oacc[nf] = vz;

  bf16_t* Pw = &Pls[wid * 16 * SPLDS];
  const int kmax = q0 + 64;
  for (int kc = 0; kc < kmax; kc += 32) {
    __syncthreads();
    // Stage K chunk via async DMA: 32 keys x 192 bf16 = 768 x 16B transfers.
#pragma unroll
    for (int it = 0; it < 6; ++it) {
      int slot = tid + it * 128;
      int r = slot / 24;
      int sg = slot - r * 24;
      async_b128_to_lds(lds_off(&Kls[r * SKLDS + sg * 8]),
                        (unsigned)(((unsigned)(kc + r) * DIM_QHD + sg * 8) * 2u),
                        Kbase);
    }
    // Stage V chunk transposed (needs per-element shuffle, stays on VALU path).
    {
      const bf16_t* vsrc = Vg + (bh + kc) * DIM_VD;
#pragma unroll 4
      for (int it = 0; it < 32; ++it) {
        int idx = tid * 32 + it;
        int vk = idx >> 7;
        int vd = idx & 127;
        Vls[vd * SVLDS + vk] = vsrc[(size_t)vk * DIM_VD + vd];
      }
    }
    wait_async0();
    __syncthreads();

    if (kc <= qw + 15) {  // wave-uniform causal chunk skip
      // scores 16x32 = 2 fragments, K-dim 192 in 6 steps
      v8f sc[2];
      sc[0] = vz; sc[1] = vz;
#pragma unroll
      for (int d = 0; d < 6; ++d) {
        v16bf kf0 = load_frag(&Kls[0 * SKLDS + d * 32], SKLDS);
        v16bf kf1 = load_frag(&Kls[16 * SKLDS + d * 32], SKLDS);
        sc[0] = wmma_bf16(qf[d], kf0, sc[0]);
        sc[1] = wmma_bf16(qf[d], kf1, sc[1]);
      }
      // causal + pad mask
#pragma unroll
      for (int f = 0; f < 2; ++f) {
        int key = kc + f * 16 + lc;
        int ok = amask[b * DIM_S + key];
#pragma unroll
        for (int e = 0; e < 8; ++e) {
          int row = qw + rh + e;
          if (key > row || ok <= 0) sc[f][e] = -1e30f;
        }
      }
      // row max across the 16-lane half
      float rmx[8];
#pragma unroll
      for (int e = 0; e < 8; ++e) rmx[e] = fmaxf(sc[0][e], sc[1][e]);
#pragma unroll
      for (int m = 8; m >= 1; m >>= 1)
#pragma unroll
        for (int e = 0; e < 8; ++e) rmx[e] = fmaxf(rmx[e], __shfl_xor(rmx[e], m, 32));
      float alpha[8];
#pragma unroll
      for (int e = 0; e < 8; ++e) {
        float mn = fmaxf(mrow[e], rmx[e]);
        alpha[e] = __expf(mrow[e] - mn);
        mrow[e] = mn;
      }
      // p = exp(s - m); row sums; stash P (bf16) in LDS in row-major 16x32
      float rsum[8] = {0.f, 0.f, 0.f, 0.f, 0.f, 0.f, 0.f, 0.f};
#pragma unroll
      for (int f = 0; f < 2; ++f)
#pragma unroll
        for (int e = 0; e < 8; ++e) {
          float pv = __expf(sc[f][e] - mrow[e]);
          rsum[e] += pv;
          Pw[(rh + e) * SPLDS + f * 16 + lc] = (bf16_t)pv;
        }
#pragma unroll
      for (int m = 8; m >= 1; m >>= 1)
#pragma unroll
        for (int e = 0; e < 8; ++e) rsum[e] += __shfl_xor(rsum[e], m, 32);
#pragma unroll
      for (int e = 0; e < 8; ++e) lrow[e] = lrow[e] * alpha[e] + rsum[e];
#pragma unroll
      for (int nf = 0; nf < 8; ++nf)
#pragma unroll
        for (int e = 0; e < 8; ++e) oacc[nf][e] *= alpha[e];

      // make the cross-lane P stores visible before re-reading as an A-frag
      asm volatile("s_wait_dscnt 0" ::: "memory");
      v16bf pf = load_frag(Pw, SPLDS);
#pragma unroll
      for (int nf = 0; nf < 8; ++nf) {
        v16bf vf = load_frag(&Vls[(nf * 16) * SVLDS], SVLDS);
        oacc[nf] = wmma_bf16(pf, vf, oacc[nf]);
      }
    }
  }

  // Write O directly in (b, s, h*VD + d) layout as bf16 for the o-proj GEMM.
#pragma unroll
  for (int nf = 0; nf < 8; ++nf) {
    int d = nf * 16 + lc;
#pragma unroll
    for (int e = 0; e < 8; ++e) {
      int row = qw + rh + e;
      float v = oacc[nf][e] / lrow[e];
      O[((size_t)b * DIM_S + row) * (DIM_NH * DIM_VD) + h * DIM_VD + d] = (bf16_t)v;
    }
  }
}

// ===================== host launch =====================
extern "C" void kernel_launch(void* const* d_in, const int* in_sizes, int n_in,
                              void* d_out, int out_size, void* d_ws, size_t ws_size,
                              hipStream_t stream) {
  (void)in_sizes; (void)n_in; (void)out_size; (void)ws_size;
  const float* hidden    = (const float*)d_in[0];
  const float* sinT      = (const float*)d_in[1];
  const float* cosT      = (const float*)d_in[2];
  const int*   pos       = (const int*)d_in[3];
  const int*   amask     = (const int*)d_in[4];
  const float* in_ln_w   = (const float*)d_in[5];
  const float* q_a_w     = (const float*)d_in[6];
  const float* q_a_ln_w  = (const float*)d_in[7];
  const float* q_b_w     = (const float*)d_in[8];
  const float* kv_a_w    = (const float*)d_in[9];
  const float* kv_a_ln_w = (const float*)d_in[10];
  const float* kv_b_w    = (const float*)d_in[11];
  const float* o_w       = (const float*)d_in[12];
  const float* post_ln_w = (const float*)d_in[13];
  const float* gate_w    = (const float*)d_in[14];
  const float* up_w      = (const float*)d_in[15];
  const float* down_w    = (const float*)d_in[16];
  float* out = (float*)d_out;

  char* wp = (char*)d_ws;
  auto alloc = [&](size_t bytes) -> void* {
    void* r = (void*)wp;
    wp += (bytes + 255) & ~(size_t)255;
    return r;
  };
  bf16_t* xn   = (bf16_t*)alloc((size_t)ROWS * DIM_H * 2);
  float*  qa   = (float*) alloc((size_t)ROWS * DIM_QLR * 4);
  bf16_t* qan  = (bf16_t*)alloc((size_t)ROWS * DIM_QLR * 2);
  float*  qf32 = (float*) alloc((size_t)ROWS * DIM_NH * DIM_QHD * 4);
  float*  ckv  = (float*) alloc((size_t)ROWS * (DIM_KVLR + DIM_ROPE) * 4);
  bf16_t* ckvn = (bf16_t*)alloc((size_t)ROWS * DIM_KVLR * 2);
  bf16_t* kpe  = (bf16_t*)alloc((size_t)ROWS * DIM_ROPE * 2);
  float*  kvf  = (float*) alloc((size_t)ROWS * DIM_NH * 256 * 4);
  bf16_t* Qbf  = (bf16_t*)alloc((size_t)ROWS * DIM_NH * DIM_QHD * 2);
  bf16_t* Kbf  = (bf16_t*)alloc((size_t)ROWS * DIM_NH * DIM_QHD * 2);
  bf16_t* Vbf  = (bf16_t*)alloc((size_t)ROWS * DIM_NH * DIM_VD * 2);
  bf16_t* Obf  = (bf16_t*)alloc((size_t)ROWS * DIM_H * 2);
  float*  h2   = (float*) alloc((size_t)ROWS * DIM_H * 4);
  bf16_t* ybf  = (bf16_t*)alloc((size_t)ROWS * DIM_H * 2);
  float*  gat  = (float*) alloc((size_t)ROWS * DIM_I * 4);
  bf16_t* hbuf = (bf16_t*)alloc((size_t)ROWS * DIM_I * 2);

  auto gemm = [&](const bf16_t* A, const float* Bw, int M, int N, int K, int mode,
                  float* oF, bf16_t* oB, const float* res, const float* aux) {
    dim3 g((unsigned)((N + BN - 1) / BN), (unsigned)(M / BM));
    gemm_bf16_kernel<<<g, 256, 0, stream>>>(A, Bw, M, N, K, mode, oF, oB, res, aux);
  };

  const float scale = 1.0f / sqrtf((float)DIM_QHD);

  // x = rms(hidden) -> bf16
  rmsnorm_kernel<<<ROWS, 256, 0, stream>>>(hidden, DIM_H, in_ln_w, xn, DIM_H, DIM_H);
  // q_a = x @ q_a_w ; rms -> bf16 ; q = @ q_b_w
  gemm(xn, q_a_w, ROWS, DIM_QLR, DIM_H, 0, qa, nullptr, nullptr, nullptr);
  rmsnorm_kernel<<<ROWS, 256, 0, stream>>>(qa, DIM_QLR, q_a_ln_w, qan, DIM_QLR, DIM_QLR);
  gemm(qan, q_b_w, ROWS, DIM_NH * DIM_QHD, DIM_QLR, 0, qf32, nullptr, nullptr, nullptr);
  // ckv = x @ kv_a_w ; split: rms(512) -> bf16, rope(64) -> k_pe
  gemm(xn, kv_a_w, ROWS, DIM_KVLR + DIM_ROPE, DIM_H, 0, ckv, nullptr, nullptr, nullptr);
  rmsnorm_kernel<<<ROWS, 256, 0, stream>>>(ckv, DIM_KVLR + DIM_ROPE, kv_a_ln_w,
                                           ckvn, DIM_KVLR, DIM_KVLR);
  rope_k_kernel<<<ROWS, 64, 0, stream>>>(ckv, cosT, sinT, pos, kpe);
  // kv = ckv_n @ kv_b_w
  gemm(ckvn, kv_b_w, ROWS, DIM_NH * 256, DIM_KVLR, 0, kvf, nullptr, nullptr, nullptr);
  // rope q (fold SCALE), build K/V bf16 layouts
  rope_q_kernel<<<dim3(ROWS, DIM_NH), 192, 0, stream>>>(qf32, cosT, sinT, pos, Qbf, scale);
  build_kv_kernel<<<dim3(ROWS, DIM_NH), 192, 0, stream>>>(kvf, kpe, Kbf, Vbf);
  // flash attention -> Obf (b, s, h*VD) bf16
  flash_attn_kernel<<<dim3(DIM_S / 64, DIM_NH, DIM_B), 128, 0, stream>>>(
      Qbf, Kbf, Vbf, amask, Obf);
  // o-proj + residual
  gemm(Obf, o_w, ROWS, DIM_H, DIM_H, 1, h2, nullptr, hidden, nullptr);
  // MLP: y = rms(h2); gate; up fused with silu(gate)*up -> bf16; down + residual -> out
  rmsnorm_kernel<<<ROWS, 256, 0, stream>>>(h2, DIM_H, post_ln_w, ybf, DIM_H, DIM_H);
  gemm(ybf, gate_w, ROWS, DIM_I, DIM_H, 0, gat, nullptr, nullptr, nullptr);
  gemm(ybf, up_w, ROWS, DIM_I, DIM_H, 3, nullptr, hbuf, nullptr, gat);
  gemm(hbuf, down_w, ROWS, DIM_H, DIM_I, 1, out, nullptr, h2, nullptr);
}
